// SKLinear_triton_16449724745548
// MI455X (gfx1250) — matrix operand order repaired and checked by reading, not compiled, not run on hardware
//
#include <hip/hip_runtime.h>
#include <hip/hip_bf16.h>

typedef __attribute__((ext_vector_type(16))) __bf16 v16bf;
typedef __attribute__((ext_vector_type(8)))  __bf16 v8bf;
typedef __attribute__((ext_vector_type(8)))  float  v8f;
typedef __attribute__((ext_vector_type(4)))  float  v4f;
typedef __attribute__((ext_vector_type(4)))  int    v4i;

#define DIN  4096
#define DOUT 4096
#define NB   8192   // batch rows
#define KRNK 64
#define RR   512    // 2 * L * K

#define LDS_STRIDE 40          // 32 bf16 payload + 8 pad -> 20-dword bank stride
#define TILE_ELEMS (128 * LDS_STRIDE)

#if __has_builtin(__builtin_amdgcn_global_load_async_to_lds_b128)
#define USE_ASYNC_LDS 1
#else
#define USE_ASYNC_LDS 0
#endif

// 16-byte global -> LDS copy; async (ASYNCcnt-tracked) when available.
__device__ __forceinline__ void copy16_g2lds(const void* g, void* l) {
#if USE_ASYNC_LDS
  __builtin_amdgcn_global_load_async_to_lds_b128(
      (v4i*)g,
      (__attribute__((address_space(3))) v4i*)l, 0, 0);
#else
  typedef __attribute__((ext_vector_type(4))) unsigned int v4u;
  *(v4u*)l = *(const v4u*)g;
#endif
}

// ---------------------------------------------------------------------------
// pack1[j*DIN + i] = A_cat[i][j] (bf16, column-major), j = br*256 + l*64 + k
//   br==0 -> S1s[l][i][k], br==1 -> U2s[l][i][k]
// ---------------------------------------------------------------------------
__global__ void pack_stage1(const float* __restrict__ S1s,
                            const float* __restrict__ U2s,
                            __bf16* __restrict__ pack1) {
  long idx = (long)blockIdx.x * blockDim.x + threadIdx.x;   // [0, RR*DIN)
  int j = (int)(idx / DIN);
  int i = (int)(idx % DIN);
  int br = j >> 8;
  int lk = j & 255;
  int l  = lk >> 6;
  int k  = lk & 63;
  const float* src = br ? U2s : S1s;
  pack1[idx] = (__bf16)src[((long)l * DIN + i) * KRNK + k];
}

// ---------------------------------------------------------------------------
// pack2[o*RR + t] = W_cat[t][o] (bf16, column-major), t = br*256 + l*64 + k
//   br==0 -> U1s[l][k][o], br==1 -> S2s[l][k][o]
// ---------------------------------------------------------------------------
__global__ void pack_stage2(const float* __restrict__ U1s,
                            const float* __restrict__ S2s,
                            __bf16* __restrict__ pack2) {
  long idx = (long)blockIdx.x * blockDim.x + threadIdx.x;   // [0, DOUT*RR)
  int o = (int)(idx / RR);
  int t = (int)(idx % RR);
  int br = t >> 8;
  int lk = t & 255;
  int l  = lk >> 6;
  int k  = lk & 63;
  const float* src = br ? S2s : U1s;
  pack2[idx] = (__bf16)src[((long)l * KRNK + k) * DOUT + o];
}

// ---------------------------------------------------------------------------
// WMMA GEMM with double-buffered LDS tiles + async global->LDS copies.
//   C[M x N] = A[M x Kdim] * B[Kdim x N];  B column-major bf16 (stride ldb).
//   STAGE1: A f32 (h_in), converted during LDS staging; C stored bf16.
//   STAGE2: A bf16 (T), async-staged; C = 0.125*acc + bias, stored f32.
// Block: 256 thr = 8 wave32, BM=BN=128, BK=32, wave tile 32x64 (2x4 WMMA).
// ---------------------------------------------------------------------------
template<bool STAGE1>
__global__ __launch_bounds__(256)
void gemm_wmma(const void* __restrict__ Av, const __bf16* __restrict__ Bp,
               void* __restrict__ Cv, const float* __restrict__ bias,
               int Kdim, int lda, int ldb, int ldc) {
  __shared__ __bf16 smem[4 * TILE_ELEMS];   // A0,A1,B0,B1 -> 40 KB

  const int tid  = threadIdx.x;
  const int lane = tid & 31;
  const int wid  = tid >> 5;
  const int wm   = wid & 3;                 // 4 row-groups of 32
  const int wn   = wid >> 2;                // 2 col-groups of 64
  const int mblk = blockIdx.y * 128;
  const int nblk = blockIdx.x * 128;
  const int lrow = lane & 15;
  const int hi   = lane >> 4;               // K-half selector (ISA layout)

  v8f acc[2][4];
  v8f zero = {};
  #pragma unroll
  for (int a = 0; a < 2; ++a)
    #pragma unroll
    for (int b = 0; b < 4; ++b) acc[a][b] = zero;

  // Stage one BK=32 slab (A tile 128x32, B tile 128cols x 32k) into buffer s.
  auto stage = [&](int k0, int s) {
    __bf16* bA = smem + s * TILE_ELEMS;
    __bf16* bB = smem + (2 + s) * TILE_ELEMS;
    #pragma unroll
    for (int j = 0; j < 2; ++j) {
      int c   = tid * 2 + j;                // 512 16B-chunks per tile
      int row = c >> 2;                     // row (A) / column (B)
      int seg = c & 3;                      // 8-element segment within slab
      copy16_g2lds(Bp + (long)(nblk + row) * ldb + k0 + seg * 8,
                   bB + row * LDS_STRIDE + seg * 8);
      if constexpr (STAGE1) {
        const float* src = (const float*)Av + (long)(mblk + row) * lda + k0 + seg * 8;
        v4f f0 = *(const v4f*)(src);
        v4f f1 = *(const v4f*)(src + 4);
        v8bf d;
        #pragma unroll
        for (int e = 0; e < 4; ++e) { d[e] = (__bf16)f0[e]; d[4 + e] = (__bf16)f1[e]; }
        *(v8bf*)(bA + row * LDS_STRIDE + seg * 8) = d;
      } else {
        copy16_g2lds((const __bf16*)Av + (long)(mblk + row) * lda + k0 + seg * 8,
                     bA + row * LDS_STRIDE + seg * 8);
      }
    }
  };

  stage(0, 0);
  int cur = 0;
  for (int k0 = 0; k0 < Kdim; k0 += 32) {
    const bool more = (k0 + 32) < Kdim;
    if (more) stage(k0 + 32, cur ^ 1);      // prefetch next slab into other buf

#if USE_ASYNC_LDS
    constexpr int PER_ITER = STAGE1 ? 2 : 4;  // async ops issued per thread/slab
    if (more) {
#if __has_builtin(__builtin_amdgcn_s_wait_asynccnt)
      __builtin_amdgcn_s_wait_asynccnt(PER_ITER);
#else
      asm volatile("s_wait_asynccnt %0" :: "n"(PER_ITER));
#endif
    } else {
#if __has_builtin(__builtin_amdgcn_s_wait_asynccnt)
      __builtin_amdgcn_s_wait_asynccnt(0);
#else
      asm volatile("s_wait_asynccnt 0");
#endif
    }
#endif
    __syncthreads();                        // current buffer resident for all waves

    const __bf16* bA = smem + cur * TILE_ELEMS;
    const __bf16* bB = smem + (2 + cur) * TILE_ELEMS;

    // A frags: lane = row, lanes0-15 K{0..7,16..23}, lanes16-31 K{8..15,24..31}
    v16bf afrag[2];
    #pragma unroll
    for (int mt = 0; mt < 2; ++mt) {
      const __bf16* la = bA + (wm * 32 + mt * 16 + lrow) * LDS_STRIDE + hi * 8;
      v8bf a0 = *(const v8bf*)(la);
      v8bf a1 = *(const v8bf*)(la + 16);
      v16bf a;
      #pragma unroll
      for (int e = 0; e < 8; ++e) { a[e] = a0[e]; a[8 + e] = a1[e]; }
      afrag[mt] = a;
    }
    // B frags: lane = column, lanes0-15 K{0..7,16..23}, lanes16-31 K{8..15,24..31}
    v16bf bfrag[4];
    #pragma unroll
    for (int nt = 0; nt < 4; ++nt) {
      const __bf16* lb = bB + (wn * 64 + nt * 16 + lrow) * LDS_STRIDE + hi * 8;
      v8bf b0 = *(const v8bf*)(lb);
      v8bf b1 = *(const v8bf*)(lb + 16);
      v16bf b;
      #pragma unroll
      for (int e = 0; e < 8; ++e) { b[e] = b0[e]; b[8 + e] = b1[e]; }
      bfrag[nt] = b;
    }

    #pragma unroll
    for (int mt = 0; mt < 2; ++mt)
      #pragma unroll
      for (int nt = 0; nt < 4; ++nt)
        acc[mt][nt] = __builtin_amdgcn_wmma_f32_16x16x32_bf16(
            false, afrag[mt], false, bfrag[nt],
            (short)0, acc[mt][nt], false, false);

    __syncthreads();                        // reads done before buffer reuse
    cur ^= 1;
  }

  // Epilogue: C/D layout -> element e at row base+hi*8+e, col = lane&15.
  #pragma unroll
  for (int mt = 0; mt < 2; ++mt) {
    #pragma unroll
    for (int nt = 0; nt < 4; ++nt) {
      int col   = nblk + wn * 64 + nt * 16 + (lane & 15);
      int rbase = mblk + wm * 32 + mt * 16 + hi * 8;
      if constexpr (STAGE1) {
        __bf16* C = (__bf16*)Cv;
        #pragma unroll
        for (int e = 0; e < 8; ++e)
          C[(long)(rbase + e) * ldc + col] = (__bf16)acc[mt][nt][e];
      } else {
        float* C = (float*)Cv;
        float bv = bias[col];
        #pragma unroll
        for (int e = 0; e < 8; ++e)
          C[(long)(rbase + e) * ldc + col] = acc[mt][nt][e] * 0.125f + bv;
      }
    }
  }
}

// ---------------------------------------------------------------------------
extern "C" void kernel_launch(void* const* d_in, const int* in_sizes, int n_in,
                              void* d_out, int out_size, void* d_ws, size_t ws_size,
                              hipStream_t stream) {
  const float* h_in = (const float*)d_in[0];
  const float* S1s  = (const float*)d_in[1];
  const float* S2s  = (const float*)d_in[2];
  const float* U1s  = (const float*)d_in[3];
  const float* U2s  = (const float*)d_in[4];
  const float* bias = (const float*)d_in[5];

  char* ws = (char*)d_ws;
  __bf16* pack1 = (__bf16*)ws;                                   // 512*4096 bf16 (4 MB)
  __bf16* pack2 = (__bf16*)(ws + (size_t)RR * DIN * 2);          // 4096*512 bf16 (4 MB)
  __bf16* T     = (__bf16*)(ws + (size_t)RR * DIN * 2
                               + (size_t)DOUT * RR * 2);         // 8192*512 bf16 (8 MB)

  pack_stage1<<<(RR * DIN) / 256, 256, 0, stream>>>(S1s, U2s, pack1);
  pack_stage2<<<(DOUT * RR) / 256, 256, 0, stream>>>(U1s, S2s, pack2);

  // T = h_in(f32->bf16) @ A_cat        [8192 x 512]
  gemm_wmma<true><<<dim3(RR / 128, NB / 128), 256, 0, stream>>>(
      (const void*)h_in, pack1, (void*)T, nullptr, DIN, DIN, DIN, RR);

  // y = 0.125 * (T @ W_cat) + bias     [8192 x 4096]
  gemm_wmma<false><<<dim3(DOUT / 128, NB / 128), 256, 0, stream>>>(
      (const void*)T, pack2, d_out, bias, RR, RR, RR, DOUT);
}
